// Seq2SeqGenerator_87522843561633
// MI455X (gfx1250) — matrix operand order repaired and checked by reading, not compiled, not run on hardware
//
#include <hip/hip_runtime.h>
#include <hip/hip_bf16.h>

#define VOCABSZ 32000
#define EMBSZ   512
#define HIDSZ   1024
#define BATCHSZ 16
#define SEQLEN  128

typedef __attribute__((ext_vector_type(16))) __bf16 v16bf;
typedef __attribute__((ext_vector_type(8)))  __bf16 v8bf;
typedef __attribute__((ext_vector_type(4)))  __bf16 v4bf;
typedef __attribute__((ext_vector_type(8)))  float  v8f;
typedef __attribute__((ext_vector_type(4)))  int    v4i;

__device__ __forceinline__ v16bf cat8(v8bf lo, v8bf hi) {
  return __builtin_shufflevector(lo, hi, 0,1,2,3,4,5,6,7,8,9,10,11,12,13,14,15);
}

// ---------------------------------------------------------------------------
// CDNA5 async global->LDS copy (ASYNCcnt-tracked). Builtin signature (from
// compiler diagnostic): (int4 AS(1)*, int4 AS(3)*, imm offset, imm cpol).
// Guarded so we still compile on toolchains without the builtin.
// ---------------------------------------------------------------------------
#if __has_builtin(__builtin_amdgcn_global_load_async_to_lds_b128)
#define ASYNC_COPY_ENABLED 1
__device__ __forceinline__ void cp_async16(__bf16* lds, const __bf16* g) {
  __builtin_amdgcn_global_load_async_to_lds_b128(
      (__attribute__((address_space(1))) v4i*)g,
      (__attribute__((address_space(3))) v4i*)lds,
      0, 0);
}
#else
#define ASYNC_COPY_ENABLED 0
__device__ __forceinline__ void cp_async16(__bf16* lds, const __bf16* g) {
  *(v8bf*)lds = *(const v8bf*)g;   // synchronous fallback (still uses DS path)
}
#endif

__device__ __forceinline__ void cp_wait_async() {
#if ASYNC_COPY_ENABLED
#if __has_builtin(__builtin_amdgcn_s_wait_asynccnt)
  __builtin_amdgcn_s_wait_asynccnt(0);
#else
  asm volatile("s_wait_asynccnt 0x0" ::: "memory");
#endif
#endif
}

// ---------------------------------------------------------------------------
// LDS-staged GEMM: C[M,N] = A[M,K](bf16 rm) * W[N,K](bf16 rm)^T + bias
// Block: 256 thr = 8 waves arranged 4(M) x 2(N); wave tile 16x64;
// block tile 64x128. K staged in 64-wide chunks, double-buffered in LDS via
// GLOBAL_LOAD_ASYNC_TO_LDS_B128. LDS rows padded to 72 bf16 (36 DWORDs) so
// ds_load_b128 across 16 rows spreads over banks.
// Requires M%64==0, N%128==0, K%64==0.
// ---------------------------------------------------------------------------
#define KB  64
#define LDA 72
#define LDB 72

__global__ __launch_bounds__(256)
void gemm_bf16_lds(const __bf16* __restrict__ A, const __bf16* __restrict__ W,
                   const float* __restrict__ bias, float* __restrict__ C,
                   int M, int N, int K) {
  __shared__ __bf16 la[2][64 * LDA];    // 2 x 9 KB
  __shared__ __bf16 lb[2][128 * LDB];   // 2 x 18 KB

  const int tid  = threadIdx.x;
  const int w    = tid >> 5;
  const int lane = tid & 31;
  const int wm   = w >> 1;              // 0..3
  const int wn   = w & 1;               // 0..1
  const int half = lane >> 4;
  const int lid  = lane & 15;
  const int m0b  = blockIdx.y * 64;
  const int n0b  = blockIdx.x * 128;

  const int cc = (tid & 7) * 8;         // 16B chunk column (elements)
  const int cr = tid >> 3;              // base row 0..31

  v8f acc0 = {}, acc1 = {}, acc2 = {}, acc3 = {};

  auto issue = [&](int st, int kb) {
    #pragma unroll
    for (int i = 0; i < 2; ++i) {       // A tile: 64 rows x 64 cols
      const int r = cr + i * 32;
      cp_async16(&la[st][r * LDA + cc], A + (size_t)(m0b + r) * K + kb + cc);
    }
    #pragma unroll
    for (int i = 0; i < 4; ++i) {       // B tile: 128 rows x 64 cols
      const int r = cr + i * 32;
      cp_async16(&lb[st][r * LDB + cc], W + (size_t)(n0b + r) * K + kb + cc);
    }
  };

  issue(0, 0);
  const int nkb = K / KB;
  for (int kb = 0; kb < nkb; ++kb) {
    const int st = kb & 1;
    cp_wait_async();                    // stage `st` resident
    __syncthreads();
    if (kb + 1 < nkb) issue(st ^ 1, (kb + 1) * KB);
    #pragma unroll
    for (int s = 0; s < 2; ++s) {       // two 32-deep WMMA substeps
      const int ks = s * 32;
      const __bf16* ap = &la[st][(wm * 16 + lid) * LDA + ks + half * 16];
      v16bf a = cat8(*(const v8bf*)ap, *(const v8bf*)(ap + 8));
      const __bf16* b0p = &lb[st][(wn * 64 +  0 + lid) * LDB + ks + half * 16];
      const __bf16* b1p = &lb[st][(wn * 64 + 16 + lid) * LDB + ks + half * 16];
      const __bf16* b2p = &lb[st][(wn * 64 + 32 + lid) * LDB + ks + half * 16];
      const __bf16* b3p = &lb[st][(wn * 64 + 48 + lid) * LDB + ks + half * 16];
      v16bf b0 = cat8(*(const v8bf*)b0p, *(const v8bf*)(b0p + 8));
      v16bf b1 = cat8(*(const v8bf*)b1p, *(const v8bf*)(b1p + 8));
      v16bf b2 = cat8(*(const v8bf*)b2p, *(const v8bf*)(b2p + 8));
      v16bf b3 = cat8(*(const v8bf*)b3p, *(const v8bf*)(b3p + 8));
      acc0 = __builtin_amdgcn_wmma_f32_16x16x32_bf16(false, a, false, b0, (short)0, acc0, false, false);
      acc1 = __builtin_amdgcn_wmma_f32_16x16x32_bf16(false, a, false, b1, (short)0, acc1, false, false);
      acc2 = __builtin_amdgcn_wmma_f32_16x16x32_bf16(false, a, false, b2, (short)0, acc2, false, false);
      acc3 = __builtin_amdgcn_wmma_f32_16x16x32_bf16(false, a, false, b3, (short)0, acc3, false, false);
    }
    __syncthreads();                    // done reading stage `st`
  }

  const int m0 = m0b + wm * 16;
  const int n0 = n0b + wn * 64;
  const float bv0 = bias[n0 +  0 + lid];
  const float bv1 = bias[n0 + 16 + lid];
  const float bv2 = bias[n0 + 32 + lid];
  const float bv3 = bias[n0 + 48 + lid];
  float* crow = C + (size_t)(m0 + half * 8) * N + n0 + lid;
  #pragma unroll
  for (int r = 0; r < 8; ++r) {
    crow[(size_t)r * N +  0] = acc0[r] + bv0;
    crow[(size_t)r * N + 16] = acc1[r] + bv1;
    crow[(size_t)r * N + 32] = acc2[r] + bv2;
    crow[(size_t)r * N + 48] = acc3[r] + bv3;
  }
}

// ---------------------------------------------------------------------------
// Direct-from-global GEMM for the tiny recurrent step (M=16): gh = h @ W_hh^T.
// 8 waves, each a 16x64 tile; block tile 16x512; latency-bound, no LDS.
// ---------------------------------------------------------------------------
__global__ __launch_bounds__(256)
void gemm_bf16_m16(const __bf16* __restrict__ A, const __bf16* __restrict__ W,
                   const float* __restrict__ bias, float* __restrict__ C,
                   int N, int K) {
  const int w    = threadIdx.x >> 5;
  const int lane = threadIdx.x & 31;
  const int n0   = blockIdx.x * 512 + w * 64;
  const int half = lane >> 4;
  const int lid  = lane & 15;

  v8f acc0 = {}, acc1 = {}, acc2 = {}, acc3 = {};
  const __bf16* arow = A + (size_t)lid * K + half * 16;
  const __bf16* w0 = W + (size_t)(n0 +  0 + lid) * K + half * 16;
  const __bf16* w1 = W + (size_t)(n0 + 16 + lid) * K + half * 16;
  const __bf16* w2 = W + (size_t)(n0 + 32 + lid) * K + half * 16;
  const __bf16* w3 = W + (size_t)(n0 + 48 + lid) * K + half * 16;

  for (int k0 = 0; k0 < K; k0 += 32) {
    v16bf a  = cat8(*(const v8bf*)(arow + k0), *(const v8bf*)(arow + k0 + 8));
    v16bf b0 = *(const v16bf*)(w0 + k0);
    v16bf b1 = *(const v16bf*)(w1 + k0);
    v16bf b2 = *(const v16bf*)(w2 + k0);
    v16bf b3 = *(const v16bf*)(w3 + k0);
    acc0 = __builtin_amdgcn_wmma_f32_16x16x32_bf16(false, a, false, b0, (short)0, acc0, false, false);
    acc1 = __builtin_amdgcn_wmma_f32_16x16x32_bf16(false, a, false, b1, (short)0, acc1, false, false);
    acc2 = __builtin_amdgcn_wmma_f32_16x16x32_bf16(false, a, false, b2, (short)0, acc2, false, false);
    acc3 = __builtin_amdgcn_wmma_f32_16x16x32_bf16(false, a, false, b3, (short)0, acc3, false, false);
  }

  const float bv0 = bias[n0 +  0 + lid];
  const float bv1 = bias[n0 + 16 + lid];
  const float bv2 = bias[n0 + 32 + lid];
  const float bv3 = bias[n0 + 48 + lid];
  float* crow = C + (size_t)(half * 8) * N + n0 + lid;
  #pragma unroll
  for (int r = 0; r < 8; ++r) {
    crow[(size_t)r * N +  0] = acc0[r] + bv0;
    crow[(size_t)r * N + 16] = acc1[r] + bv1;
    crow[(size_t)r * N + 32] = acc2[r] + bv2;
    crow[(size_t)r * N + 48] = acc3[r] + bv3;
  }
}

// ---------------------------------------------------------------------------
__global__ __launch_bounds__(256)
void cvt_f32_bf16(const float* __restrict__ src, __bf16* __restrict__ dst, int n) {
  int i = (blockIdx.x * 256 + threadIdx.x) * 4;
  if (i >= n) return;
  float4 f = *(const float4*)(src + i);
  v4bf o;
  o[0] = (__bf16)f.x; o[1] = (__bf16)f.y; o[2] = (__bf16)f.z; o[3] = (__bf16)f.w;
  *(v4bf*)(dst + i) = o;
}

__global__ __launch_bounds__(256)
void embed_gather(const int* __restrict__ x, const float* __restrict__ emb,
                  __bf16* __restrict__ enc_in, __bf16* __restrict__ dec_in) {
  int idx = blockIdx.x * 256 + threadIdx.x;        // over B*T*(E/4)
  int e4  = idx % (EMBSZ / 4);
  int bt  = idx / (EMBSZ / 4);
  int tok = x[bt];
  float4 f = *(const float4*)(emb + (size_t)tok * EMBSZ + e4 * 4);
  v4bf v;
  v[0] = (__bf16)f.x; v[1] = (__bf16)f.y; v[2] = (__bf16)f.z; v[3] = (__bf16)f.w;
  *(v4bf*)(enc_in + (size_t)bt * EMBSZ + e4 * 4) = v;
  v4bf dv = v;
  if ((bt % SEQLEN) == 0) {
    float4 f0 = *(const float4*)(emb + e4 * 4);    // row START_IDX=0
    dv[0] = (__bf16)f0.x; dv[1] = (__bf16)f0.y; dv[2] = (__bf16)f0.z; dv[3] = (__bf16)f0.w;
  }
  *(v4bf*)(dec_in + (size_t)bt * EMBSZ + e4 * 4) = dv;
}

__global__ __launch_bounds__(256)
void zero_h(float* __restrict__ h, __bf16* __restrict__ h_bf) {
  int idx = blockIdx.x * 256 + threadIdx.x;
  h[idx] = 0.0f;
  h_bf[idx] = (__bf16)0.0f;
}

// GRU gate fusion (PyTorch gate order r,z,n):
//   r = sig(gi_r + gh_r); z = sig(gi_z + gh_z); n = tanh(gi_n + r*gh_n)
//   h' = (1-z)*n + z*h
__global__ __launch_bounds__(256)
void gru_gate(const float* __restrict__ gi, const float* __restrict__ gh,
              float* __restrict__ h, __bf16* __restrict__ h_bf,
              __bf16* __restrict__ out_bf, int t) {
  int idx = blockIdx.x * 256 + threadIdx.x;        // B*H
  int b = idx >> 10;
  int j = idx & (HIDSZ - 1);
  const float* git = gi + ((size_t)(b * SEQLEN + t)) * (3 * HIDSZ);
  const float* ghb = gh + (size_t)b * (3 * HIDSZ);
  float ir = git[j],             hr = ghb[j];
  float iz = git[HIDSZ + j],     hz = ghb[HIDSZ + j];
  float in = git[2 * HIDSZ + j], hn = ghb[2 * HIDSZ + j];
  float r = 1.0f / (1.0f + __expf(-(ir + hr)));
  float z = 1.0f / (1.0f + __expf(-(iz + hz)));
  float n = tanhf(in + r * hn);
  float hv = h[idx];
  float hN = (1.0f - z) * n + z * hv;
  h[idx] = hN;
  h_bf[idx] = (__bf16)hN;
  if (out_bf) out_bf[((size_t)(b * SEQLEN + t)) * HIDSZ + j] = (__bf16)hN;
}

// ---------------------------------------------------------------------------
extern "C" void kernel_launch(void* const* d_in, const int* in_sizes, int n_in,
                              void* d_out, int out_size, void* d_ws, size_t ws_size,
                              hipStream_t stream) {
  (void)in_sizes; (void)n_in; (void)out_size; (void)ws_size;
  const int*   x    = (const int*)  d_in[0];
  const float* emb  = (const float*)d_in[1];
  const float* Wihe = (const float*)d_in[2];
  const float* Whhe = (const float*)d_in[3];
  const float* bihe = (const float*)d_in[4];
  const float* bhhe = (const float*)d_in[5];
  const float* Wihd = (const float*)d_in[6];
  const float* Whhd = (const float*)d_in[7];
  const float* bihd = (const float*)d_in[8];
  const float* bhhd = (const float*)d_in[9];
  const float* Wfc  = (const float*)d_in[10];
  const float* bfc  = (const float*)d_in[11];
  float* out = (float*)d_out;

  const int BT = BATCHSZ * SEQLEN;                 // 2048
  char* ws = (char*)d_ws;
  size_t off = 0;
  auto alloc = [&](size_t bytes) -> char* {
    char* p = ws + off;
    off += (bytes + 255) & ~(size_t)255;
    return p;
  };
  __bf16* enc_in     = (__bf16*)alloc((size_t)BT * EMBSZ * 2);
  __bf16* dec_in     = (__bf16*)alloc((size_t)BT * EMBSZ * 2);
  __bf16* Wihe_bf    = (__bf16*)alloc((size_t)3 * HIDSZ * EMBSZ * 2);
  __bf16* Whhe_bf    = (__bf16*)alloc((size_t)3 * HIDSZ * HIDSZ * 2);
  __bf16* Wihd_bf    = (__bf16*)alloc((size_t)3 * HIDSZ * EMBSZ * 2);
  __bf16* Whhd_bf    = (__bf16*)alloc((size_t)3 * HIDSZ * HIDSZ * 2);
  __bf16* Wfc_bf     = (__bf16*)alloc((size_t)VOCABSZ * HIDSZ * 2);
  float*  gi         = (float*) alloc((size_t)BT * 3 * HIDSZ * 4);
  float*  gh         = (float*) alloc((size_t)BATCHSZ * 3 * HIDSZ * 4);
  float*  h          = (float*) alloc((size_t)BATCHSZ * HIDSZ * 4);
  __bf16* h_bf       = (__bf16*)alloc((size_t)BATCHSZ * HIDSZ * 2);
  __bf16* dec_out_bf = (__bf16*)alloc((size_t)BT * HIDSZ * 2);

  auto cvt = [&](const float* s, __bf16* d, int n) {
    cvt_f32_bf16<<<(n / 4 + 255) / 256, 256, 0, stream>>>(s, d, n);
  };
  cvt(Wihe, Wihe_bf, 3 * HIDSZ * EMBSZ);
  cvt(Whhe, Whhe_bf, 3 * HIDSZ * HIDSZ);
  cvt(Wihd, Wihd_bf, 3 * HIDSZ * EMBSZ);
  cvt(Whhd, Whhd_bf, 3 * HIDSZ * HIDSZ);
  cvt(Wfc,  Wfc_bf,  VOCABSZ * HIDSZ);

  embed_gather<<<(BT * (EMBSZ / 4)) / 256, 256, 0, stream>>>(x, emb, enc_in, dec_in);

  // Encoder input projections: gi = enc_in @ W_ih^T + b_ih   [2048, 3072]
  gemm_bf16_lds<<<dim3(3 * HIDSZ / 128, BT / 64), 256, 0, stream>>>(
      enc_in, Wihe_bf, bihe, gi, BT, 3 * HIDSZ, EMBSZ);
  zero_h<<<(BATCHSZ * HIDSZ) / 256, 256, 0, stream>>>(h, h_bf);
  for (int t = 0; t < SEQLEN; ++t) {
    gemm_bf16_m16<<<dim3(3 * HIDSZ / 512), 256, 0, stream>>>(
        h_bf, Whhe_bf, bhhe, gh, 3 * HIDSZ, HIDSZ);
    gru_gate<<<(BATCHSZ * HIDSZ) / 256, 256, 0, stream>>>(gi, gh, h, h_bf, nullptr, t);
  }

  // Decoder: reuse gi buffer; h currently holds h_enc (decoder h0).
  gemm_bf16_lds<<<dim3(3 * HIDSZ / 128, BT / 64), 256, 0, stream>>>(
      dec_in, Wihd_bf, bihd, gi, BT, 3 * HIDSZ, EMBSZ);
  for (int t = 0; t < SEQLEN; ++t) {
    gemm_bf16_m16<<<dim3(3 * HIDSZ / 512), 256, 0, stream>>>(
        h_bf, Whhd_bf, bhhd, gh, 3 * HIDSZ, HIDSZ);
    gru_gate<<<(BATCHSZ * HIDSZ) / 256, 256, 0, stream>>>(gi, gh, h, h_bf, dec_out_bf, t);
  }

  // Final FC: logits[2048, 32000] = dec_out @ W_fc^T + b_fc  (bf16 W_fc is L2-resident)
  gemm_bf16_lds<<<dim3(VOCABSZ / 128, BT / 64), 256, 0, stream>>>(
      dec_out_bf, Wfc_bf, bfc, out, BT, VOCABSZ, HIDSZ);

  // hidden output: [1, B, H] appended after logits
  (void)hipMemcpyAsync(out + (size_t)BT * VOCABSZ, h,
                       (size_t)BATCHSZ * HIDSZ * sizeof(float),
                       hipMemcpyDeviceToDevice, stream);
}
